// ProtoCoxLoss_74680891343472
// MI455X (gfx1250) — matrix operand order repaired and checked by reading, not compiled, not run on hardware
//
#include <hip/hip_runtime.h>

#define TAU      0.07f
#define EPS_NORM 1e-12f
#define EPS_LOG  1e-8f
#define TILE_J   4096

typedef __attribute__((ext_vector_type(16))) __bf16 v16bf;
typedef __attribute__((ext_vector_type(8)))  float  v8f;

static __device__ __forceinline__ unsigned short f2bf_rne(float x) {
    unsigned u = __float_as_uint(x);
    unsigned r = ((u >> 16) & 1u) + 0x7FFFu;     // round-to-nearest-even
    return (unsigned short)((u + r) >> 16);
}
static __device__ __forceinline__ float bf2f(unsigned short h) {
    return __uint_as_float(((unsigned)h) << 16);
}
static __device__ __forceinline__ unsigned maskpk(float t0, float t1, float ti) {
    return (t0 >= ti ? 0x00003F80u : 0u) |       // bf16 1.0 in low half
           (t1 >= ti ? 0x3F800000u : 0u);        // bf16 1.0 in high half
}

// Phase 1: logits[i] = (f_i . p) / (max(|f_i|,eps)*max(|p|,eps)*tau);
// e = exp(logit) split into two bf16 terms e_hi + e_lo (~f32-class when re-summed).
__global__ void __launch_bounds__(256)
k_prep(const float* __restrict__ feats, const float* __restrict__ proto,
       float* __restrict__ logits, unsigned short* __restrict__ ehi,
       unsigned short* __restrict__ elo, int N, int D) {
    const int lane   = threadIdx.x & 31;
    const int wave   = (blockIdx.x * blockDim.x + threadIdx.x) >> 5;
    const int nwaves = (gridDim.x * blockDim.x) >> 5;
    const int d4     = lane * 4;

    float4 p4 = make_float4(0.f, 0.f, 0.f, 0.f);
    if (d4 + 3 < D) p4 = *(const float4*)(proto + d4);
    float pss = p4.x*p4.x + p4.y*p4.y + p4.z*p4.z + p4.w*p4.w;
    #pragma unroll
    for (int o = 16; o > 0; o >>= 1) pss += __shfl_xor(pss, o, 32);
    const float inv_pt = 1.0f / (fmaxf(sqrtf(pss), EPS_NORM) * TAU);

    for (int r = wave; r < N; r += nwaves) {
        float4 f4 = make_float4(0.f, 0.f, 0.f, 0.f);
        if (d4 + 3 < D) f4 = *(const float4*)(feats + (size_t)r * D + d4);
        float ss = f4.x*f4.x + f4.y*f4.y + f4.z*f4.z + f4.w*f4.w;
        float dp = f4.x*p4.x + f4.y*p4.y + f4.z*p4.z + f4.w*p4.w;
        #pragma unroll
        for (int o = 16; o > 0; o >>= 1) {
            ss += __shfl_xor(ss, o, 32);
            dp += __shfl_xor(dp, o, 32);
        }
        float lg = dp / fmaxf(sqrtf(ss), EPS_NORM) * inv_pt;
        float e  = expf(lg);
        if (lane == 0) {
            logits[r] = lg;
            unsigned short h = f2bf_rne(e);
            ehi[r] = h;
            elo[r] = f2bf_rne(e - bf2f(h));
        }
    }
}

// Phase 2: denom[i] = sum_j [times[j] >= times[i]] * e[j], via bf16 WMMA.
// A = 16x32 bf16 0/1 mask tile (exact); B = e_{hi,lo} replicated across columns.
// Mask-gen VALU co-executes with v_wmma (XDL ops are TRANS-class).
// Unrolled x2 with independent operand buffers + 4 accumulators to avoid
// WMMA WAR-hazard NOPs and accumulator dependence chains.
__global__ void __launch_bounds__(256)
k_denom(const float* __restrict__ times, const unsigned short* __restrict__ ehi,
        const unsigned short* __restrict__ elo, float* __restrict__ denom, int N) {
    __shared__ __align__(16) float    s_t[TILE_J];
    __shared__ __align__(16) unsigned s_eh[TILE_J / 2];
    __shared__ __align__(16) unsigned s_el[TILE_J / 2];

    const int tid   = threadIdx.x;
    const int lane  = tid & 31;
    const int wv    = tid >> 5;                       // 8 waves, one 16-row i-tile each
    const int i0    = blockIdx.x * 128 + wv * 16;
    const int m     = lane & 15;                      // A row (and D row group)
    const int koff  = (lane >> 4) << 3;               // A: lanes>=16 hold K+8
    const int kbase = (lane >> 4) << 3;               // B: lanes>=16 hold K+8 (same as A)
    const float ti  = times[i0 + m];

    const unsigned* gh = (const unsigned*)ehi;        // 2 bf16 per dword
    const unsigned* gl = (const unsigned*)elo;

    v8f accH0 = {}, accL0 = {}, accH1 = {}, accL1 = {};

    for (int jt = 0; jt < N; jt += TILE_J) {
        __syncthreads();
        for (int x = tid * 4; x < TILE_J; x += 256 * 4)
            *(float4*)&s_t[x] = *(const float4*)&times[jt + x];
        for (int x = tid * 4; x < TILE_J / 2; x += 256 * 4) {
            *(uint4*)&s_eh[x] = *(const uint4*)&gh[(jt >> 1) + x];
            *(uint4*)&s_el[x] = *(const uint4*)&gl[(jt >> 1) + x];
        }
        __syncthreads();

        for (int jb = 0; jb < TILE_J; jb += 64) {     // uniform: EXEC all-1s at WMMA
            // ---- chunk 0: j in [jb, jb+32) ----
            float4 a00 = *(const float4*)&s_t[jb + koff];
            float4 a01 = *(const float4*)&s_t[jb + koff + 4];
            float4 a02 = *(const float4*)&s_t[jb + 16 + koff];
            float4 a03 = *(const float4*)&s_t[jb + 16 + koff + 4];
            // ---- chunk 1: j in [jb+32, jb+64) ----
            float4 a10 = *(const float4*)&s_t[jb + 32 + koff];
            float4 a11 = *(const float4*)&s_t[jb + 32 + koff + 4];
            float4 a12 = *(const float4*)&s_t[jb + 48 + koff];
            float4 a13 = *(const float4*)&s_t[jb + 48 + koff + 4];

            union { unsigned u[8]; v16bf v; } A0, BH0, BL0, A1, BH1, BL1;
            const int ub0 = (jb + kbase) >> 1;             // dword idx, 16B aligned
            const int ub1 = (jb + 32 + kbase) >> 1;
            *(uint4*)&BH0.u[0] = *(const uint4*)&s_eh[ub0];
            *(uint4*)&BH0.u[4] = *(const uint4*)&s_eh[ub0 + 8];
            *(uint4*)&BL0.u[0] = *(const uint4*)&s_el[ub0];
            *(uint4*)&BL0.u[4] = *(const uint4*)&s_el[ub0 + 8];
            *(uint4*)&BH1.u[0] = *(const uint4*)&s_eh[ub1];
            *(uint4*)&BH1.u[4] = *(const uint4*)&s_eh[ub1 + 8];
            *(uint4*)&BL1.u[0] = *(const uint4*)&s_el[ub1];
            *(uint4*)&BL1.u[4] = *(const uint4*)&s_el[ub1 + 8];

            A0.u[0] = maskpk(a00.x, a00.y, ti); A0.u[1] = maskpk(a00.z, a00.w, ti);
            A0.u[2] = maskpk(a01.x, a01.y, ti); A0.u[3] = maskpk(a01.z, a01.w, ti);
            A0.u[4] = maskpk(a02.x, a02.y, ti); A0.u[5] = maskpk(a02.z, a02.w, ti);
            A0.u[6] = maskpk(a03.x, a03.y, ti); A0.u[7] = maskpk(a03.z, a03.w, ti);

            accH0 = __builtin_amdgcn_wmma_f32_16x16x32_bf16(
                        false, A0.v, false, BH0.v, (short)0, accH0, false, false);
            accL0 = __builtin_amdgcn_wmma_f32_16x16x32_bf16(
                        false, A0.v, false, BL0.v, (short)0, accL0, false, false);

            A1.u[0] = maskpk(a10.x, a10.y, ti); A1.u[1] = maskpk(a10.z, a10.w, ti);
            A1.u[2] = maskpk(a11.x, a11.y, ti); A1.u[3] = maskpk(a11.z, a11.w, ti);
            A1.u[4] = maskpk(a12.x, a12.y, ti); A1.u[5] = maskpk(a12.z, a12.w, ti);
            A1.u[6] = maskpk(a13.x, a13.y, ti); A1.u[7] = maskpk(a13.z, a13.w, ti);

            accH1 = __builtin_amdgcn_wmma_f32_16x16x32_bf16(
                        false, A1.v, false, BH1.v, (short)0, accH1, false, false);
            accL1 = __builtin_amdgcn_wmma_f32_16x16x32_bf16(
                        false, A1.v, false, BL1.v, (short)0, accL1, false, false);
        }
    }

    v8f acc = (accH0 + accL0) + (accH1 + accL1);
    // D layout: lane 0 holds M=0..7 (N=0), lane 16 holds M=8..15 (N=0)
    if (lane == 0) {
        #pragma unroll
        for (int r = 0; r < 8; ++r) denom[i0 + r] = acc[r];
    } else if (lane == 16) {
        #pragma unroll
        for (int r = 0; r < 8; ++r) denom[i0 + 8 + r] = acc[r];
    }
}

// Phase 3a: per-block partial sums of numerator / normalizer (deterministic tree).
__global__ void __launch_bounds__(256)
k_loss_part(const float* __restrict__ logits, const float* __restrict__ denom,
            const int* __restrict__ events, const float* __restrict__ w,
            float* __restrict__ partials, int N) {
    __shared__ float sn[256], sd[256];
    const int tid = threadIdx.x;
    float num = 0.f, den = 0.f;
    for (int i = blockIdx.x * blockDim.x + tid; i < N; i += gridDim.x * blockDim.x) {
        float ew = (float)events[i] * w[i];
        float ld = logf(denom[i] + EPS_LOG);
        num += -(logits[i] - ld) * ew;
        den += ew;
    }
    sn[tid] = num; sd[tid] = den;
    __syncthreads();
    for (int s = 128; s > 0; s >>= 1) {
        if (tid < s) { sn[tid] += sn[tid + s]; sd[tid] += sd[tid + s]; }
        __syncthreads();
    }
    if (tid == 0) {
        partials[2 * blockIdx.x]     = sn[0];
        partials[2 * blockIdx.x + 1] = sd[0];
    }
}

// Phase 3b: fixed-order final combine.
__global__ void k_loss_final(const float* __restrict__ partials, float* __restrict__ out,
                             int nb) {
    if (threadIdx.x == 0 && blockIdx.x == 0) {
        float num = 0.f, den = 0.f;
        for (int b = 0; b < nb; ++b) { num += partials[2 * b]; den += partials[2 * b + 1]; }
        out[0] = num / (den + EPS_LOG);
    }
}

extern "C" void kernel_launch(void* const* d_in, const int* in_sizes, int n_in,
                              void* d_out, int out_size, void* d_ws, size_t ws_size,
                              hipStream_t stream) {
    const float* feats  = (const float*)d_in[0];
    const float* times  = (const float*)d_in[1];
    const int*   events = (const int*)d_in[2];
    const float* w      = (const float*)d_in[3];
    const float* proto  = (const float*)d_in[4];
    const int N = in_sizes[1];
    const int D = in_sizes[4];

    float* ws = (float*)d_ws;
    float*          logits   = ws;                                    // N floats
    float*          denom    = ws + N;                                // N floats
    unsigned short* ehi      = (unsigned short*)(ws + 2 * N);         // N bf16
    unsigned short* elo      = (unsigned short*)(ws + 2 * N + N / 2); // N bf16
    float*          partials = ws + 3 * N;                            // 2*nb floats

    k_prep<<<128, 256, 0, stream>>>(feats, proto, logits, ehi, elo, N, D);
    k_denom<<<N / 128, 256, 0, stream>>>(times, ehi, elo, denom, N);
    const int nb = 64;
    k_loss_part<<<nb, 256, 0, stream>>>(logits, denom, events, w, partials, N);
    k_loss_final<<<1, 32, 0, stream>>>(partials, (float*)d_out, nb);
}
